// GatingNetwork_82411832475900
// MI455X (gfx1250) — matrix operand order repaired and checked by reading, not compiled, not run on hardware
//
#include <hip/hip_runtime.h>
#include <hip/hip_bf16.h>
#include <math.h>

// CDNA5 / gfx1250, wave32. WMMA bf16 16x16x32, fp32 accumulate.
// Fused gating network: single pass over X (roofline floor: 134MB @ 23.3TB/s),
// fp32->bf16 in registers, row L2-norm fused into the GEMM K-loop.

typedef __bf16 bf16_t;
typedef __bf16 v16bf __attribute__((ext_vector_type(16)));
typedef float  v8f   __attribute__((ext_vector_type(8)));
typedef float  v4f   __attribute__((ext_vector_type(4)));

// ---------------------------------------------------------------------------
// Kernel 1: column-L2-normalize sim_matrix [C x E] and store transposed
// (expert-major, [E][C]) as bf16 into workspace. 256 KB -> L2 resident.
// ---------------------------------------------------------------------------
__global__ __launch_bounds__(256)
void norm_experts_kernel(const float* __restrict__ S, bf16_t* __restrict__ Bn,
                         int C, int E) {
    const int e = blockIdx.x;           // one block per expert column
    __shared__ float red[256];
    float p = 0.f;
    for (int k = threadIdx.x; k < C; k += 256) {
        float v = S[(size_t)k * E + e];
        p += v * v;
    }
    red[threadIdx.x] = p;
    __syncthreads();
    for (int s = 128; s > 0; s >>= 1) {
        if ((int)threadIdx.x < s) red[threadIdx.x] += red[threadIdx.x + s];
        __syncthreads();
    }
    const float inv = 1.0f / fmaxf(sqrtf(red[0]), 1e-12f);
    for (int k = threadIdx.x; k < C; k += 256) {
        Bn[(size_t)e * C + k] = (bf16_t)(S[(size_t)k * E + e] * inv);
    }
}

// ---------------------------------------------------------------------------
// Kernel 2: one wave32 per 32-row tile (two 16-row M-tiles sharing B frags),
// 64 experts = 4 WMMA column tiles -> 8 accumulators.
// A fragments built directly from global (4x b128 per lane per tile, same
// coalescing as an LDS-staged scheme but zero DS traffic / no loop barrier).
// ---------------------------------------------------------------------------
__global__ __launch_bounds__(32)
void gating_kernel(const float* __restrict__ X, const bf16_t* __restrict__ Bn,
                   const float* __restrict__ gates, const int* __restrict__ fkp,
                   float* __restrict__ out, int N, int C) {
    const int L    = threadIdx.x;          // lane 0..31
    const int row0 = blockIdx.x * 32;

    __shared__ float dotLDS[32 * 64];      // raw dot products (8 KB)
    __shared__ float ssqLDS[64];           // per-(tile,lane) sum of squares
    __shared__ float gLDS[64];             // gates

    gLDS[L]      = gates[L];
    gLDS[L + 32] = gates[L + 32];

    // 16-bit WMMA A layout (contiguous, matching B): lane L -> row m = L&15,
    // elements i=0..15 at K = akb+i,
    // with akb = 0 for lanes 0-15, 16 for lanes 16-31.
    const int m   = L & 15;
    const int akb = (L >> 4) * 16;
    const int bkb = (L >> 4) * 16;         // B: lanes 0-15 K=0-15, 16-31 K=16-31

    const float* xr0 = X + (size_t)(row0 + m)      * C + akb;  // M-tile 0
    const float* xr1 = X + (size_t)(row0 + 16 + m) * C + akb;  // M-tile 1

    v8f acc[2][4];
    #pragma unroll
    for (int t = 0; t < 2; ++t)
        #pragma unroll
        for (int n = 0; n < 4; ++n) acc[t][n] = (v8f){};

    float ssq0 = 0.f, ssq1 = 0.f;

    for (int k0 = 0; k0 < C; k0 += 32) {
        // A tile 0: rows row0+0..15
        v4f a0q0 = *(const v4f*)(xr0 + k0);
        v4f a0q1 = *(const v4f*)(xr0 + k0 + 4);
        v4f a0q2 = *(const v4f*)(xr0 + k0 + 8);
        v4f a0q3 = *(const v4f*)(xr0 + k0 + 12);
        // A tile 1: rows row0+16..31
        v4f a1q0 = *(const v4f*)(xr1 + k0);
        v4f a1q1 = *(const v4f*)(xr1 + k0 + 4);
        v4f a1q2 = *(const v4f*)(xr1 + k0 + 8);
        v4f a1q3 = *(const v4f*)(xr1 + k0 + 12);
        // B fragments: expert (n*16+m), 16 contiguous bf16 (32B) per lane
        const bf16_t* bp = Bn + k0 + bkb;
        v16bf b0 = *(const v16bf*)(bp + (size_t)( 0 + m) * C);
        v16bf b1 = *(const v16bf*)(bp + (size_t)(16 + m) * C);
        v16bf b2 = *(const v16bf*)(bp + (size_t)(32 + m) * C);
        v16bf b3 = *(const v16bf*)(bp + (size_t)(48 + m) * C);

        if (k0 + 32 < C) {                 // global_prefetch_b8 for next chunk
            __builtin_prefetch(xr0 + k0 + 32, 0, 3);
            __builtin_prefetch(xr1 + k0 + 32, 0, 3);
        }

        // convert to bf16 fragments + fuse row sum-of-squares
        v16bf a0, a1;
        #pragma unroll
        for (int j = 0; j < 4; ++j) {
            ssq0 += a0q0[j]*a0q0[j] + a0q1[j]*a0q1[j] + a0q2[j]*a0q2[j] + a0q3[j]*a0q3[j];
            ssq1 += a1q0[j]*a1q0[j] + a1q1[j]*a1q1[j] + a1q2[j]*a1q2[j] + a1q3[j]*a1q3[j];
            a0[j]      = (bf16_t)a0q0[j];
            a0[4 + j]  = (bf16_t)a0q1[j];
            a0[8 + j]  = (bf16_t)a0q2[j];
            a0[12 + j] = (bf16_t)a0q3[j];
            a1[j]      = (bf16_t)a1q0[j];
            a1[4 + j]  = (bf16_t)a1q1[j];
            a1[8 + j]  = (bf16_t)a1q2[j];
            a1[12 + j] = (bf16_t)a1q3[j];
        }

        acc[0][0] = __builtin_amdgcn_wmma_f32_16x16x32_bf16(false, a0, false, b0, (short)0, acc[0][0], false, false);
        acc[0][1] = __builtin_amdgcn_wmma_f32_16x16x32_bf16(false, a0, false, b1, (short)0, acc[0][1], false, false);
        acc[0][2] = __builtin_amdgcn_wmma_f32_16x16x32_bf16(false, a0, false, b2, (short)0, acc[0][2], false, false);
        acc[0][3] = __builtin_amdgcn_wmma_f32_16x16x32_bf16(false, a0, false, b3, (short)0, acc[0][3], false, false);
        acc[1][0] = __builtin_amdgcn_wmma_f32_16x16x32_bf16(false, a1, false, b0, (short)0, acc[1][0], false, false);
        acc[1][1] = __builtin_amdgcn_wmma_f32_16x16x32_bf16(false, a1, false, b1, (short)0, acc[1][1], false, false);
        acc[1][2] = __builtin_amdgcn_wmma_f32_16x16x32_bf16(false, a1, false, b2, (short)0, acc[1][2], false, false);
        acc[1][3] = __builtin_amdgcn_wmma_f32_16x16x32_bf16(false, a1, false, b3, (short)0, acc[1][3], false, false);
    }

    // dump C fragments: VGPR g of lane L holds (M = g + (L<16?0:8), N = L&15)
    ssqLDS[L]      = ssq0;
    ssqLDS[L + 32] = ssq1;
    const int mrow = (L >> 4) * 8;
    #pragma unroll
    for (int g = 0; g < 8; ++g) {
        #pragma unroll
        for (int n = 0; n < 4; ++n) {
            dotLDS[(mrow + g) * 64      + n * 16 + m] = acc[0][n][g];
            dotLDS[(mrow + g + 16) * 64 + n * 16 + m] = acc[1][n][g];
        }
    }
    __syncthreads();

    // per-row epilogue: all 32 lanes, one row of 64 experts each
    {
        const int r   = L;
        const int row = row0 + r;
        // row r's sum-of-squares halves: lanes (r&15) and (r&15)+16 of its tile
        const int  sb  = (r >> 4) * 32 + (r & 15);
        const float s  = ssqLDS[sb] + ssqLDS[sb + 16];
        const float inv = 1.0f / fmaxf(sqrtf(s), 1e-12f);
        const int   fk  = *fkp;                 // fallback_k (2 in reference)
        const float* d  = &dotLDS[r * 64];

        // pass 1: count active, top-2 of raw logits (first-occurrence ties)
        float v1 = -INFINITY, v2 = -INFINITY;
        int   i1 = -1, i2 = -1, nact = 0;
        for (int e = 0; e < 64; ++e) {
            float lg = d[e] * inv - gLDS[e];
            if (lg > 0.f) ++nact;
            if (lg > v1)      { v2 = v1; i2 = i1; v1 = lg; i1 = e; }
            else if (lg > v2) { v2 = lg; i2 = e; }
        }
        const bool  fb = (nact == 0);
        // masked values are gated=relu(logits); fallback rows have gated==0
        const float mx = fb ? 0.f : v1;

        float sum = 0.f;
        for (int e = 0; e < 64; ++e) {
            float lg = d[e] * inv - gLDS[e];
            bool act = fb ? (e == i1 || (fk > 1 && e == i2)) : (lg > 0.f);
            if (act) sum += __expf(fmaxf(lg, 0.f) - mx);
        }
        const float rs = 1.0f / sum;

        const size_t o  = (size_t)row * 64;
        const size_t NE = (size_t)N * 64;
        for (int e = 0; e < 64; ++e) {
            float lg = d[e] * inv - gLDS[e];
            bool act = fb ? (e == i1 || (fk > 1 && e == i2)) : (lg > 0.f);
            out[o + e]          = act ? __expf(fmaxf(lg, 0.f) - mx) * rs : 0.f;
            out[NE + o + e]     = lg;
            out[2 * NE + o + e] = act ? 1.0f : 0.f;
        }
    }
}

// ---------------------------------------------------------------------------
extern "C" void kernel_launch(void* const* d_in, const int* in_sizes, int n_in,
                              void* d_out, int out_size, void* d_ws, size_t ws_size,
                              hipStream_t stream) {
    const float* X     = (const float*)d_in[0];   // hidden_states [b,t,C]
    const float* S     = (const float*)d_in[1];   // sim_matrix    [C,E]
    const float* gates = (const float*)d_in[2];   // [E]
    const int*   fk    = (const int*)d_in[3];     // fallback_k

    const int E = in_sizes[2];                    // 64
    const int C = in_sizes[1] / E;                // 2048
    const int N = in_sizes[0] / C;                // 16384

    bf16_t* Bn = (bf16_t*)d_ws;                   // normalized experts, [E][C] bf16

    norm_experts_kernel<<<E, 256, 0, stream>>>(S, Bn, C, E);
    gating_kernel<<<N / 32, 32, 0, stream>>>(X, Bn, gates, fk, (float*)d_out, N, C);
}
